// SpikingDenseLayer_78357383348568
// MI455X (gfx1250) — matrix operand order repaired and checked by reading, not compiled, not run on hardware
//
#include <hip/hip_runtime.h>

// ---------------------------------------------------------------------------
// Spiking dense layer for MI455X (gfx1250, wave32, WMMA f16 16x16x32).
//   1) prep:    w -> f16, pre-swizzled into WMMA-B fragment order + inv_norm
//   2) dmat:    d = w^T w, f16, WMMA-B fragment order
//   3) hgemm:   h = x @ w (f16 WMMA, f32 acc); B staged via async global->LDS
//   4) synscan: in-place h -> syn_t (linear IIR, 65536 parallel chains)
//   5) scan:    serial 512-step membrane recurrence, rst = spk @ d via WMMA;
//               spikes in LDS (A-frag order), mem in VGPRs (C-frag order).
// Workspace: [0,2MiB) w_frag | [2,4MiB) d_frag | [4MiB) inv_norm | [8MiB) h/syn.
// ---------------------------------------------------------------------------

typedef __attribute__((ext_vector_type(16))) _Float16 v16h;
typedef __attribute__((ext_vector_type(8)))  float    v8f;
typedef int v4i __attribute__((vector_size(16)));    // matches builtin param type

#define IN_C   1024
#define OUT_C  1024
#define TSTEPS 512
#define ALPHA_F 0.8187307530779818f  // exp(-0.001/0.005)
#define BETA_F  0.9048374180359595f  // exp(-0.001/0.01)
#define EPS_F   1e-8f

#if __has_builtin(__builtin_amdgcn_global_load_async_to_lds_b128)
#define HAVE_ASYNC_LDS 1
// global param is addrspace(1) v4i*, LDS param is addrspace(3) v4i*
#define GPTR(p) ((__attribute__((address_space(1))) v4i*)(const void*)(p))
#define LPTR(p) ((__attribute__((address_space(3))) v4i*)(void*)(p))
#endif

// --- WMMA fragment index maps (CDNA5 ISA §7.12.2, wave32) -------
__device__ __forceinline__ int laneA(int m, int kk) { return m + 16*((kk>>3)&1); }
__device__ __forceinline__ int jA(int kk)           { return 2*(4*(kk>>4) + ((kk&7)>>1)) + (kk&1); }
__device__ __forceinline__ int laneB(int kk, int nn) { return nn + 16*((kk>>3)&1); }
__device__ __forceinline__ int jB(int kk)            { return 8*(kk>>4) + (kk&7); }
// B-fragment-order global position for element (k, n) of a KxN f16 matrix
// tiled 32(K) x 16(N), tiles row-major over (k>>5, n>>4), N-tiles per row = 64.
__device__ __forceinline__ size_t fragPos(int k, int n) {
  return ((size_t)((k >> 5) * 64 + (n >> 4)) * 32 + laneB(k & 31, n & 15)) * 16
         + jB(k & 31);
}

// --- 1) w -> f16 in B-fragment order + column norms ------------------------
__global__ void prep_kernel(const float* __restrict__ w,
                            _Float16* __restrict__ wfrag,
                            float* __restrict__ inv_norm) {
  int j = blockIdx.x * 256 + threadIdx.x;          // 4 blocks x 256 = 1024 cols
  float acc = 0.f;
  for (int a = 0; a < IN_C; ++a) {
    float v = w[a * OUT_C + j];
    wfrag[fragPos(a, j)] = (_Float16)v;
    acc += v * v;
  }
  inv_norm[j] = 1.0f / (acc + EPS_F);
}

// --- 2) d = w^T w, output f16 in WMMA-B fragment order ---------------------
__global__ __launch_bounds__(256) void dmat_kernel(const float* __restrict__ w,
                                                   _Float16* __restrict__ dfrag) {
  __shared__ float wi[16][64];
  __shared__ float wj[16][64];
  int tx = threadIdx.x, ty = threadIdx.y;
  int i0 = blockIdx.x * 64, j0 = blockIdx.y * 64;
  float acc[4][4] = {};
  for (int k0 = 0; k0 < IN_C; k0 += 16) {
    __syncthreads();
    for (int u = 0; u < 4; ++u) {
      wi[ty][tx + 16*u] = w[(k0 + ty) * OUT_C + i0 + tx + 16*u];
      wj[ty][tx + 16*u] = w[(k0 + ty) * OUT_C + j0 + tx + 16*u];
    }
    __syncthreads();
    for (int kk = 0; kk < 16; ++kk)
      for (int a4 = 0; a4 < 4; ++a4)
        for (int b4 = 0; b4 < 4; ++b4)
          acc[a4][b4] += wi[kk][ty + 16*a4] * wj[kk][tx + 16*b4];
  }
  for (int a4 = 0; a4 < 4; ++a4)
    for (int b4 = 0; b4 < 4; ++b4) {
      int i = i0 + ty + 16*a4;                     // d row == K index of rst GEMM
      int j = j0 + tx + 16*b4;                     // d col == N index
      dfrag[fragPos(i, j)] = (_Float16)acc[a4][b4];
    }
}

// --- 3) h = x @ w  (f16 WMMA, 128x128 block, 8 waves of 64x32) -------------
__global__ __launch_bounds__(256) void hgemm_kernel(const float* __restrict__ x,
                                                    const _Float16* __restrict__ wfrag,
                                                    float* __restrict__ h) {
  __shared__ _Float16 ldsA[8 * 512];               // 8 M-tiles, fragment order
  __shared__ _Float16 ldsB[8 * 512];               // 8 N-tiles, fragment order
  int t = threadIdx.x;
  int lane = t & 31, wid = t >> 5;
  int wm = wid & 1, wn = wid >> 1;                 // 2x4 wave grid
  int m0 = blockIdx.y * 128, n0 = blockIdx.x * 128;
  v8f acc[4][2] = {};
  for (int k0 = 0; k0 < IN_C; k0 += 32) {
    __syncthreads();
    { // stage B: contiguous 8KB fragment-ordered tile, async global->LDS
      const _Float16* src =
          wfrag + (size_t)((k0 >> 5) * 64 + (n0 >> 4)) * 512 + t * 16;
#ifdef HAVE_ASYNC_LDS
      __builtin_amdgcn_global_load_async_to_lds_b128(GPTR(src),     LPTR(ldsB + t*16),     0, 0);
      __builtin_amdgcn_global_load_async_to_lds_b128(GPTR(src + 8), LPTR(ldsB + t*16 + 8), 0, 0);
#else
      *(v16h*)(ldsB + t * 16) = *(const v16h*)src;
#endif
    }
    { // stage A: x f32 -> f16, packed-pair scatter into fragment order
      int r  = t >> 3;
      int kq = (t & 7) * 4;
      for (int rr = 0; rr < 4; ++rr) {
        int m = r + 32 * rr;
        const float4 v = *(const float4*)(x + (size_t)(m0 + m) * IN_C + k0 + kq);
        float vv[4] = {v.x, v.y, v.z, v.w};
        for (int u = 0; u < 4; u += 2) {
          int k = kq + u;                          // jA(k) even, jA(k+1)=jA(k)+1
          union { _Float16 h2[2]; unsigned u32; } pk;
          pk.h2[0] = (_Float16)vv[u];
          pk.h2[1] = (_Float16)vv[u + 1];
          int idx = (m >> 4) * 512 + laneA(m & 15, k) * 16 + jA(k);
          ((unsigned*)ldsA)[idx >> 1] = pk.u32;
        }
      }
    }
#ifdef HAVE_ASYNC_LDS
    asm volatile("s_wait_asynccnt 0x0" ::: "memory");
#endif
    __syncthreads();
    v16h a[4], b[2];
    for (int mt = 0; mt < 4; ++mt) a[mt] = ((const v16h*)ldsA)[(wm*4 + mt) * 32 + lane];
    for (int nt = 0; nt < 2; ++nt) b[nt] = ((const v16h*)ldsB)[(wn*2 + nt) * 32 + lane];
    for (int mt = 0; mt < 4; ++mt)
      for (int nt = 0; nt < 2; ++nt)
        acc[mt][nt] = __builtin_amdgcn_wmma_f32_16x16x32_f16(
            false, a[mt], false, b[nt], (short)0, acc[mt][nt], false, false);
  }
  for (int mt = 0; mt < 4; ++mt)
    for (int nt = 0; nt < 2; ++nt)
      for (int r = 0; r < 8; ++r) {
        int m = m0 + wm*64 + mt*16 + r + 8*(lane >> 4);
        int n = n0 + wn*32 + nt*16 + (lane & 15);
        h[(size_t)m * OUT_C + n] = acc[mt][nt][r];
      }
}

// --- 4) in-place h -> syn_t (carry value *entering* step t) ----------------
__global__ void synscan_kernel(float* __restrict__ h) {
  int g = blockIdx.x * 256 + threadIdx.x;          // 65536 chains
  int b = g >> 10, o = g & 1023;
  float* p = h + ((size_t)b * TSTEPS) * OUT_C + o;
  float syn = 0.f;
  for (int t = 0; t < TSTEPS; ++t) {
    float tmp = p[(size_t)t * OUT_C];
    p[(size_t)t * OUT_C] = syn;                    // store pre-update syn
    syn = ALPHA_F * syn + tmp;
  }
}

// --- 5) serial membrane recurrence: 4 WGs x 16 batch rows ------------------
__global__ __launch_bounds__(256) void scan_kernel(const float* __restrict__ syn,
                                                   const _Float16* __restrict__ dfrag,
                                                   const float* __restrict__ inv_norm,
                                                   const float* __restrict__ bvec,
                                                   float* __restrict__ out) {
  __shared__ _Float16 spkA[32 * 512];              // 32 K-tiles, A-fragment order
  int t = threadIdx.x;
  int lane = t & 31, wid = t >> 5;                 // wave owns N in [128w,128w+128)
  int bbase = blockIdx.x * 16;
  int m_lo = 8 * (lane >> 4);                      // C-fragment row base (0 or 8)
  int ncol = lane & 15;
  float mem[8][8] = {};                            // mem in VGPRs, C-frag layout
  float invn[8], bb[8];
  for (int nt = 0; nt < 8; ++nt) {
    int n = wid * 128 + nt * 16 + ncol;
    invn[nt] = inv_norm[n];
    bb[nt]   = bvec[n];
  }
  for (int ts = 0; ts < TSTEPS; ++ts) {
    // phase 1: spikes from mem -> global out + LDS (A-fragment order)
    for (int nt = 0; nt < 8; ++nt) {
      int n = wid * 128 + nt * 16 + ncol;
      for (int r = 0; r < 8; ++r) {
        int m = m_lo + r;
        float s = (mem[nt][r] * invn[nt] - bb[nt] > 0.f) ? 1.f : 0.f;
        out[(((size_t)(bbase + m)) * TSTEPS + ts) * OUT_C + n] = s;
        spkA[(n >> 5) * 512 + laneA(m, n & 31) * 16 + jA(n & 31)] = (_Float16)s;
      }
    }
    __syncthreads();
    // phase 2: rst = spk @ d for this wave's 8 N-tiles (256 WMMAs/wave/step)
    v8f acc[8] = {};
    for (int kt = 0; kt < 32; ++kt) {
      v16h a = ((const v16h*)spkA)[kt * 32 + lane];
      __builtin_prefetch(dfrag + ((size_t)((kt + 1) * 64 + wid * 8) * 32 + lane) * 16, 0, 1);
      for (int nt = 0; nt < 8; ++nt) {
        v16h b = *(const v16h*)(dfrag +
                  ((size_t)(kt * 64 + wid * 8 + nt) * 32 + lane) * 16);
        acc[nt] = __builtin_amdgcn_wmma_f32_16x16x32_f16(
            false, a, false, b, (short)0, acc[nt], false, false);
      }
    }
    // phase 3: mem update (pure register math + streamed syn_t)
    for (int nt = 0; nt < 8; ++nt) {
      int n = wid * 128 + nt * 16 + ncol;
      for (int r = 0; r < 8; ++r) {
        int m = m_lo + r;
        float s = syn[(((size_t)(bbase + m)) * TSTEPS + ts) * OUT_C + n];
        mem[nt][r] = BETA_F * mem[nt][r] + s - acc[nt][r];
      }
    }
    __syncthreads();                               // spkA reusable next step
  }
}

extern "C" void kernel_launch(void* const* d_in, const int* in_sizes, int n_in,
                              void* d_out, int out_size, void* d_ws, size_t ws_size,
                              hipStream_t stream) {
  const float* x = (const float*)d_in[0];          // [64,512,1024]
  const float* w = (const float*)d_in[1];          // [1024,1024]
  const float* b = (const float*)d_in[2];          // [1024]
  float* out = (float*)d_out;                      // [64,512,1024]
  char* ws = (char*)d_ws;
  _Float16* wfrag = (_Float16*)(ws);               // 2 MiB (B-fragment order)
  _Float16* dfrag = (_Float16*)(ws + (2u << 20));  // 2 MiB (B-fragment order)
  float*    inv   = (float*)(ws + (4u << 20));     // 4 KiB
  float*    h     = (float*)(ws + (8u << 20));     // 128 MiB (h, then syn)

  prep_kernel   <<<4, 256, 0, stream>>>(w, wfrag, inv);
  dmat_kernel   <<<dim3(16, 16), dim3(16, 16), 0, stream>>>(w, dfrag);
  hgemm_kernel  <<<dim3(8, 256), 256, 0, stream>>>(x, wfrag, h);
  synscan_kernel<<<256, 256, 0, stream>>>(h);
  scan_kernel   <<<4, 256, 0, stream>>>(h, dfrag, inv, b, out);
}
